// QuotientGraphVAE_30253749633529
// MI455X (gfx1250) — matrix operand order repaired and checked by reading, not compiled, not run on hardware
//
#include <hip/hip_runtime.h>
#include <hip/hip_bf16.h>

#define NN      50000      // nodes
#define NE      800000     // edges
#define ETOT    850000     // edges + self loops
#define HID     128
#define EFEAT   64
#define LAT     64

// output layout (floats)
#define OFF_NF  0          // 6400
#define OFF_EL  6400       // 2500
#define OFF_NN  8900       // 50
#define OFF_CP  8950       // 6
#define OFF_MU  8956       // 64
#define OFF_LV  9020       // 64

typedef __attribute__((ext_vector_type(16))) _Float16 v16h;
typedef __attribute__((ext_vector_type(8)))  _Float16 v8h;
typedef __attribute__((ext_vector_type(8)))  float    v8f;

// ---- order-preserving float<->uint encoding for atomic max --------------
__device__ inline unsigned enc_f(float f) {
  unsigned u = __float_as_uint(f);
  return (u & 0x80000000u) ? ~u : (u | 0x80000000u);
}
__device__ inline float dec_f(unsigned u) {
  return (u & 0x80000000u) ? __uint_as_float(u & 0x7fffffffu) : __uint_as_float(~u);
}

// ---- generic zero -------------------------------------------------------
__global__ void zero_kernel(float* p, int n) {
  int i = blockIdx.x * blockDim.x + threadIdx.x;
  if (i < n) p[i] = 0.f;
}

// ---- h = X @ W  (X:[*,128] f32, W:[128,128] f32) via f16 WMMA -----------
// block = 256 threads (8 waves); each block: 16 rows x 128 cols.
// LDS: W transposed (sWt[n][k]) so B-fragments are contiguous wide ds loads;
//      A tile staged once as f16 so all 8 waves share it.
__global__ __launch_bounds__(256) void node_gemm_kernel(
    const float* __restrict__ X, const float* __restrict__ W, float* __restrict__ H) {
  __shared__ __attribute__((aligned(32))) _Float16 sWt[128 * 128];  // 32 KB, [n][k]
  __shared__ __attribute__((aligned(32))) _Float16 sX[16 * 128];    // 4 KB,  [m][k]
  const int tid  = threadIdx.x;
  const int row0 = blockIdx.x * 16;
  for (int i = tid; i < 128 * 128; i += 256) {
    int k = i >> 7, n = i & 127;                   // coalesced read of W[k][n]
    sWt[n * 128 + k] = (_Float16)W[i];
  }
  for (int i = tid; i < 16 * 128; i += 256)
    sX[i] = (_Float16)X[(size_t)row0 * 128 + i];
  __syncthreads();

  const int lane = tid & 31;
  const int wave = tid >> 5;
  const int n    = (wave << 4) + (lane & 15);      // output column
  const int am   = lane & 15;                      // A row in lane
  const int ak   = (lane & 16) ? 16 : 0;           // A k-half
  const int bk   = (lane & 16) ? 16 : 0;           // B k-half
  v8f c = {};
#pragma unroll
  for (int k0 = 0; k0 < 128; k0 += 32) {
    v8h alo = *(const v8h*)(sX + am * 128 + k0 + ak);
    v8h ahi = *(const v8h*)(sX + am * 128 + k0 + ak + 8);
    v16h a  = __builtin_shufflevector(alo, ahi, 0, 1, 2, 3, 4, 5, 6, 7,
                                      8, 9, 10, 11, 12, 13, 14, 15);
    v16h b  = *(const v16h*)(sWt + n * 128 + k0 + bk);
    c = __builtin_amdgcn_wmma_f32_16x16x32_f16(false, a, false, b, (short)0, c, false, false);
  }
  const int mb = row0 + ((lane & 16) ? 8 : 0);
#pragma unroll
  for (int r = 0; r < 8; ++r) H[(size_t)(mb + r) * 128 + n] = c[r];
}

// ---- fused relu(EF @ We + be) column-sum (EF:[NE,64]) via WMMA ----------
__global__ __launch_bounds__(256) void edge_emb_colsum_kernel(
    const float* __restrict__ EF, const float* __restrict__ We,
    const float* __restrict__ be, float* __restrict__ esum) {
  __shared__ __attribute__((aligned(32))) _Float16 sWt[128 * 64];   // 16 KB, [n][k]
  __shared__ __attribute__((aligned(32))) _Float16 sX[16 * 64];     // 2 KB,  [m][k]
  const int tid  = threadIdx.x;
  const int row0 = blockIdx.x * 16;
  for (int i = tid; i < 64 * 128; i += 256) {
    int k = i >> 7, n = i & 127;                   // We is [64][128] row-major
    sWt[n * 64 + k] = (_Float16)We[i];
  }
  for (int i = tid; i < 16 * 64; i += 256)
    sX[i] = (_Float16)EF[(size_t)row0 * 64 + i];
  __syncthreads();

  const int lane = tid & 31;
  const int wave = tid >> 5;
  const int n    = (wave << 4) + (lane & 15);
  const int am   = lane & 15;
  const int ak   = (lane & 16) ? 16 : 0;
  const int bk   = (lane & 16) ? 16 : 0;
  v8f c = {};
#pragma unroll
  for (int k0 = 0; k0 < 64; k0 += 32) {
    v8h alo = *(const v8h*)(sX + am * 64 + k0 + ak);
    v8h ahi = *(const v8h*)(sX + am * 64 + k0 + ak + 8);
    v16h a  = __builtin_shufflevector(alo, ahi, 0, 1, 2, 3, 4, 5, 6, 7,
                                      8, 9, 10, 11, 12, 13, 14, 15);
    v16h b  = *(const v16h*)(sWt + n * 64 + k0 + bk);
    c = __builtin_amdgcn_wmma_f32_16x16x32_f16(false, a, false, b, (short)0, c, false, false);
  }
  const float bn = be[n];
  float s = 0.f;
#pragma unroll
  for (int r = 0; r < 8; ++r) { float v = c[r] + bn; s += (v > 0.f) ? v : 0.f; }
  s += __shfl_xor(s, 16, 32);                      // combine row halves (same n)
  if (!(lane & 16)) atomicAdd(&esum[n], s);
}

// ---- alpha_src/dst = h . a  (one wave per node) -------------------------
__global__ void alpha_kernel(const float* __restrict__ H,
                             const float* __restrict__ a_src, const float* __restrict__ a_dst,
                             float* __restrict__ as_out, float* __restrict__ ad_out, int n) {
  int wid  = (blockIdx.x * blockDim.x + threadIdx.x) >> 5;
  int lane = threadIdx.x & 31;
  if (wid >= n) return;
  float4 h4 = ((const float4*)(H + (size_t)wid * 128))[lane];
  float4 av = ((const float4*)a_src)[lane];
  float4 dv = ((const float4*)a_dst)[lane];
  float ss = h4.x * av.x + h4.y * av.y + h4.z * av.z + h4.w * av.w;
  float sd = h4.x * dv.x + h4.y * dv.y + h4.z * dv.z + h4.w * dv.w;
  for (int off = 16; off; off >>= 1) {
    ss += __shfl_xor(ss, off, 32);
    sd += __shfl_xor(sd, off, 32);
  }
  if (lane == 0) { as_out[wid] = ss; ad_out[wid] = sd; }
}

// ---- edge pass 1: e = leaky_relu(as[src]+ad[dst]); segment max ----------
__global__ void edge_score_kernel(const int* __restrict__ ei,
                                  const float* __restrict__ as, const float* __restrict__ ad,
                                  float* __restrict__ ebuf, unsigned* __restrict__ emax) {
  int e = blockIdx.x * blockDim.x + threadIdx.x;
  if (e >= ETOT) return;
  int s, d;
  if (e < NE) { s = ei[e]; d = ei[NE + e]; } else { s = d = e - NE; }
  float v = as[s] + ad[d];
  v = (v > 0.f) ? v : 0.2f * v;
  ebuf[e] = v;
  atomicMax(&emax[d], enc_f(v));
}

// ---- edge pass 2: e_exp = exp(e - max[dst]); segment sum ----------------
__global__ void edge_exp_kernel(const int* __restrict__ ei, float* __restrict__ ebuf,
                                const unsigned* __restrict__ emax, float* __restrict__ denom) {
  int e = blockIdx.x * blockDim.x + threadIdx.x;
  if (e >= ETOT) return;
  int d = (e < NE) ? ei[NE + e] : (e - NE);
  float x = __expf(ebuf[e] - dec_f(emax[d]));
  ebuf[e] = x;
  atomicAdd(&denom[d], x);
}

// ---- scatter: out[dst] += alpha * h[src] (one wave per edge) ------------
__global__ void scatter_kernel(const int* __restrict__ ei, const float* __restrict__ H,
                               const float* __restrict__ ebuf, const float* __restrict__ denom,
                               float* __restrict__ out) {
  int wid  = (blockIdx.x * blockDim.x + threadIdx.x) >> 5;
  int lane = threadIdx.x & 31;
  if (wid >= ETOT) return;
  int s, d;
  if (wid < NE) { s = ei[wid]; d = ei[NE + wid]; } else { s = d = wid - NE; }
  float alpha = ebuf[wid] / denom[d];
  float4 h4 = ((const float4*)(H + (size_t)s * 128))[lane];
  float* o = out + (size_t)d * 128 + lane * 4;
  atomicAdd(o + 0, alpha * h4.x);
  atomicAdd(o + 1, alpha * h4.y);
  atomicAdd(o + 2, alpha * h4.z);
  atomicAdd(o + 3, alpha * h4.w);
}

// ---- x = relu(x + bias) -------------------------------------------------
__global__ void bias_relu_kernel(float* __restrict__ x, const float* __restrict__ b, int total) {
  int i = blockIdx.x * blockDim.x + threadIdx.x;
  if (i < total) { float v = x[i] + b[i & 127]; x[i] = (v > 0.f) ? v : 0.f; }
}

// ---- column sums of x2 [NN,128] -----------------------------------------
__global__ void colsum_kernel(const float* __restrict__ X, float* __restrict__ sums, int nrows) {
  int col  = threadIdx.x & 127;
  int half = threadIdx.x >> 7;
  int rpb  = nrows / gridDim.x;
  int r0   = blockIdx.x * rpb;
  float s = 0.f;
  for (int r = r0 + half; r < r0 + rpb; r += 2) s += X[(size_t)r * 128 + col];
  atomicAdd(&sums[col], s);
}

// ---- fused tiny decoder head (single block, 256 threads) ----------------
__global__ __launch_bounds__(256) void head_kernel(
    const float* xsum, const float* esum,
    const float* Wc, const float* bc, const float* Wmu, const float* bmu,
    const float* Wlv, const float* blv, const float* eps,
    const float* Wl2h, const float* bl2h, const float* Wnh, const float* bnh,
    const float* Weh, const float* beh, const float* Wnn, const float* bnn,
    const float* Wcp, const float* bcp,
    float* out, float* node_h, float* edge_h) {
  __shared__ float cat[256], g[128], z[64], hd[128];
  int t = threadIdx.x;
  cat[t] = (t < 128) ? xsum[t] * (1.f / (float)NN) : esum[t - 128] * (1.f / (float)NE);
  __syncthreads();
  if (t < 128) {
    float s = bc[t];
    for (int k = 0; k < 256; ++k) s += cat[k] * Wc[k * 128 + t];
    g[t] = (s > 0.f) ? s : 0.f;
  }
  __syncthreads();
  if (t < 64) {
    float sm = bmu[t], sl = blv[t];
    for (int k = 0; k < 128; ++k) { float gv = g[k]; sm += gv * Wmu[k * 64 + t]; sl += gv * Wlv[k * 64 + t]; }
    out[OFF_MU + t] = sm;
    out[OFF_LV + t] = sl;
    z[t] = sm + eps[t] * __expf(0.5f * sl);
  }
  __syncthreads();
  if (t < 128) {
    float s = bl2h[t];
    for (int k = 0; k < 64; ++k) s += z[k] * Wl2h[k * 128 + t];
    hd[t] = (s > 0.f) ? s : 0.f;
  }
  __syncthreads();
  if (t < 128) {
    float s = bnh[t];
    for (int k = 0; k < 128; ++k) s += hd[k] * Wnh[k * 128 + t];
    node_h[t] = (s > 0.f) ? s : 0.f;
  } else {
    int j = t - 128;
    float s = beh[j];
    for (int k = 0; k < 128; ++k) s += hd[k] * Weh[k * 128 + j];
    edge_h[j] = (s > 0.f) ? s : 0.f;
  }
  if (t < 50) {
    float s = bnn[t];
    for (int k = 0; k < 128; ++k) s += hd[k] * Wnn[k * 50 + t];
    out[OFF_NN + t] = s;
  }
  if (t >= 64 && t < 70) {
    int j = t - 64;
    float s = bcp[j];
    for (int k = 0; k < 128; ++k) s += hd[k] * Wcp[k * 6 + j];
    out[OFF_CP + j] = s;
  }
}

// ---- batch-1 projection: out[j] = h . W[:,j] + b[j] ---------------------
__global__ void proj_kernel(const float* __restrict__ hvec, const float* __restrict__ Wbig,
                            const float* __restrict__ bbig, float* __restrict__ outp, int ncols) {
  __shared__ float sh[128];
  int t = threadIdx.x;
  if (t < 128) sh[t] = hvec[t];
  __syncthreads();
  int j = blockIdx.x * blockDim.x + t;
  if (j < ncols) {
    float s = bbig[j];
    for (int k = 0; k < 128; ++k) s += sh[k] * Wbig[(size_t)k * ncols + j];
    outp[j] = s;
  }
}

extern "C" void kernel_launch(void* const* d_in, const int* in_sizes, int n_in,
                              void* d_out, int out_size, void* d_ws, size_t ws_size,
                              hipStream_t stream) {
  const float* node_features = (const float*)d_in[0];
  const int*   edge_index    = (const int*)d_in[1];
  const float* edge_features = (const float*)d_in[2];
  const float* eps    = (const float*)d_in[3];
  const float* W1     = (const float*)d_in[4];
  const float* a_src1 = (const float*)d_in[5];
  const float* a_dst1 = (const float*)d_in[6];
  const float* b1     = (const float*)d_in[7];
  const float* W2     = (const float*)d_in[8];
  const float* a_src2 = (const float*)d_in[9];
  const float* a_dst2 = (const float*)d_in[10];
  const float* b2     = (const float*)d_in[11];
  const float* We     = (const float*)d_in[12];
  const float* be     = (const float*)d_in[13];
  const float* Wc     = (const float*)d_in[14];
  const float* bc     = (const float*)d_in[15];
  const float* Wmu    = (const float*)d_in[16];
  const float* bmu    = (const float*)d_in[17];
  const float* Wlv    = (const float*)d_in[18];
  const float* blv    = (const float*)d_in[19];
  const float* Wl2h   = (const float*)d_in[20];
  const float* bl2h   = (const float*)d_in[21];
  const float* Wnh    = (const float*)d_in[22];
  const float* bnh    = (const float*)d_in[23];
  const float* Wnf    = (const float*)d_in[24];
  const float* bnf    = (const float*)d_in[25];
  const float* Weh    = (const float*)d_in[26];
  const float* beh    = (const float*)d_in[27];
  const float* Wee    = (const float*)d_in[28];
  const float* bee    = (const float*)d_in[29];
  const float* Wnn    = (const float*)d_in[30];
  const float* bnn    = (const float*)d_in[31];
  const float* Wcp    = (const float*)d_in[32];
  const float* bcp    = (const float*)d_in[33];
  float* out = (float*)d_out;

  // workspace layout (floats)
  float* ws      = (float*)d_ws;
  float* buf_h   = ws;                                   // NN*128
  float* buf_out = buf_h + (size_t)NN * 128;             // NN*128 (x after finalize)
  float* a_s     = buf_out + (size_t)NN * 128;           // NN
  float* a_d     = a_s + NN;                             // NN
  float* emaxf   = a_d + NN;                             // NN (as unsigned)
  float* denom   = emaxf + NN;                           // NN
  float* ebuf    = denom + NN;                           // ETOT
  float* xsum    = ebuf + ETOT;                          // 128
  float* esum    = xsum + 128;                           // 128
  float* node_h  = esum + 128;                           // 128
  float* edge_h  = node_h + 128;                         // 128

  const int TB = 256;
  const int totalF = NN * 128;

  // ------------------ GAT layer 1: in = node_features ------------------
  node_gemm_kernel<<<NN / 16, TB, 0, stream>>>(node_features, W1, buf_h);
  alpha_kernel<<<NN / 8, TB, 0, stream>>>(buf_h, a_src1, a_dst1, a_s, a_d, NN);
  zero_kernel<<<(2 * NN + TB - 1) / TB, TB, 0, stream>>>(emaxf, 2 * NN);  // emax + denom
  edge_score_kernel<<<(ETOT + TB - 1) / TB, TB, 0, stream>>>(edge_index, a_s, a_d, ebuf, (unsigned*)emaxf);
  edge_exp_kernel<<<(ETOT + TB - 1) / TB, TB, 0, stream>>>(edge_index, ebuf, (const unsigned*)emaxf, denom);
  zero_kernel<<<(totalF + TB - 1) / TB, TB, 0, stream>>>(buf_out, totalF);
  scatter_kernel<<<ETOT / 8, TB, 0, stream>>>(edge_index, buf_h, ebuf, denom, buf_out);
  bias_relu_kernel<<<(totalF + TB - 1) / TB, TB, 0, stream>>>(buf_out, b1, totalF);

  // ------------------ GAT layer 2: in = buf_out (x1) ------------------
  node_gemm_kernel<<<NN / 16, TB, 0, stream>>>(buf_out, W2, buf_h);
  alpha_kernel<<<NN / 8, TB, 0, stream>>>(buf_h, a_src2, a_dst2, a_s, a_d, NN);
  zero_kernel<<<(2 * NN + TB - 1) / TB, TB, 0, stream>>>(emaxf, 2 * NN);
  edge_score_kernel<<<(ETOT + TB - 1) / TB, TB, 0, stream>>>(edge_index, a_s, a_d, ebuf, (unsigned*)emaxf);
  edge_exp_kernel<<<(ETOT + TB - 1) / TB, TB, 0, stream>>>(edge_index, ebuf, (const unsigned*)emaxf, denom);
  zero_kernel<<<(totalF + TB - 1) / TB, TB, 0, stream>>>(buf_out, totalF);
  scatter_kernel<<<ETOT / 8, TB, 0, stream>>>(edge_index, buf_h, ebuf, denom, buf_out);
  bias_relu_kernel<<<(totalF + TB - 1) / TB, TB, 0, stream>>>(buf_out, b2, totalF);

  // ------------------ graph embedding ------------------
  zero_kernel<<<1, TB, 0, stream>>>(xsum, 256);          // xsum + esum
  colsum_kernel<<<250, TB, 0, stream>>>(buf_out, xsum, NN);
  edge_emb_colsum_kernel<<<NE / 16, TB, 0, stream>>>(edge_features, We, be, esum);

  // ------------------ VAE head + decoder ------------------
  head_kernel<<<1, TB, 0, stream>>>(xsum, esum, Wc, bc, Wmu, bmu, Wlv, blv, eps,
                                    Wl2h, bl2h, Wnh, bnh, Weh, beh, Wnn, bnn, Wcp, bcp,
                                    out, node_h, edge_h);
  proj_kernel<<<25, TB, 0, stream>>>(node_h, Wnf, bnf, out + OFF_NF, 6400);
  proj_kernel<<<10, TB, 0, stream>>>(edge_h, Wee, bee, out + OFF_EL, 2500);

  (void)in_sizes; (void)n_in; (void)out_size; (void)ws_size;
}
